// Net_46368466927789
// MI455X (gfx1250) — matrix operand order repaired and checked by reading, not compiled, not run on hardware
//
#include <hip/hip_runtime.h>
#include <hip/hip_bf16.h>
#include <cstdint>
#include <cstddef>

#define SEQ   2048
#define BATCH 128
#define HID   256
#define WPAD  264   // padded bf16 row stride: row*528B -> bank offset 4*row, no LDS bank conflicts

typedef __attribute__((ext_vector_type(16))) __bf16 v16bf;
typedef __attribute__((ext_vector_type(8)))  float  v8f;

union Frag { uint4 q[2]; v16bf v; };

__device__ __forceinline__ unsigned short f2bf(float f) {
  __bf16 b = (__bf16)f;
  return __builtin_bit_cast(unsigned short, b);
}

// Hardware tanh (V_TANH_F32, TRANS32 op on gfx1250): single instruction,
// branch-free, co-executes with VALU. Avoids libm's exec-mask branchy expansion
// on the serial scan critical path.
__device__ __forceinline__ float htanh(float v) {
#if __has_builtin(__builtin_amdgcn_tanhf)
  return __builtin_amdgcn_tanhf(v);
#else
  float r;
  asm volatile("v_tanh_f32 %0, %1" : "=v"(r) : "v"(v));
  return r;
#endif
}

// Persistent RNN scan. One block = 16 batch rows (one WMMA M-tile), 8 waves,
// each wave owns 32 hidden columns. W_eff bf16 B-fragments are register-resident.
__launch_bounds__(256, 1)
__global__ void rnn_scan_kernel(const float* __restrict__ x,
                                const float* __restrict__ h0,
                                const float* __restrict__ noise,
                                const float* __restrict__ W_ih,
                                const float* __restrict__ W_hh,
                                const float* __restrict__ W_hh_bias,
                                const float* __restrict__ b_h,
                                const int*   __restrict__ ctx,
                                float* __restrict__ outp)   // (BATCH, SEQ, HID)
{
  extern __shared__ char smem[];
  unsigned short* Wb = (unsigned short*)smem;                  // [HID][WPAD] bf16 W_eff
  unsigned short* hA = (unsigned short*)(smem + HID*WPAD*2);   // [2][16][WPAD] bf16 h (double buffered)

  const int tid  = threadIdx.x;
  const int wave = tid >> 5;
  const int lane = tid & 31;
  const int b0   = blockIdx.x * 16;
  const float cf = (float)ctx[0];

  // Stage W_eff = W_hh + cf*W_hh_bias into LDS as bf16, row-major [j][k] padded.
  for (int i = tid; i < HID*HID; i += 256) {
    const int j = i >> 8, k = i & (HID - 1);
    Wb[j*WPAD + k] = f2bf(W_hh[i] + cf * W_hh_bias[i]);
  }
  // Stage this block's h0 tile into buffer 0 (row-major bf16 == A-fragment layout source).
  for (int i = tid; i < 16*HID; i += 256) {
    const int m = i >> 8, k = i & (HID - 1);
    hA[m*WPAD + k] = f2bf(h0[(b0 + m)*HID + k]);
  }
  __syncthreads();

  // Per-lane geometry (ISA 7.12.2 fragment layouts, wave32).
  const int n0   = wave*32 + (lane & 15);   // N-tile 0 column
  const int n1   = n0 + 16;                 // N-tile 1 column
  const int moff = (lane >> 4) * 8;         // C-fragment rows: v -> M = moff+v
  const int kadd = (lane >= 16) ? 16 : 0;   // B frag: lanes 16-31 hold K upper half
  const int aoff = (lane >= 16) ? 16 : 0;   // A frag: lanes 16-31 hold K upper half

  const float wi0 = W_ih[n0], wi1 = W_ih[n1];
  const float bh0 = b_h[n0],  bh1 = b_h[n1];

  // Preload all 16 B fragments of W_eff^T for this wave's 32 columns; keep in VGPRs.
  Frag B0[8], B1[8];
  #pragma unroll
  for (int kf = 0; kf < 8; ++kf) {
    const unsigned short* r0 = &Wb[n0*WPAD + kf*32 + kadd];
    const unsigned short* r1 = &Wb[n1*WPAD + kf*32 + kadd];
    B0[kf].q[0] = *(const uint4*)(r0);  B0[kf].q[1] = *(const uint4*)(r0 + 8);
    B1[kf].q[0] = *(const uint4*)(r1);  B1[kf].q[1] = *(const uint4*)(r1 + 8);
  }

  const float* xcol = x + b0 + moff;   // x[t][b0+moff+v] = xcol[t*BATCH + v]

  for (int t = 0; t < SEQ; ++t) {
    const unsigned short* hRd = hA + (t & 1)       * (16*WPAD);
    unsigned short*       hWr = hA + ((t + 1) & 1) * (16*WPAD);

    // Streaming operands for this step (overlap with WMMA below).
    float xs[8], nz0[8], nz1[8];
    size_t obase[8];
    #pragma unroll
    for (int v = 0; v < 8; ++v) {
      xs[v] = xcol[(size_t)t * BATCH + v];
      const size_t base = ((size_t)(b0 + moff + v) * SEQ + t) * HID;
      obase[v] = base;
      nz0[v] = __builtin_nontemporal_load(noise + base + n0);
      nz1[v] = __builtin_nontemporal_load(noise + base + n1);
      if (t + 1 < SEQ) {                       // global_prefetch_b8 next step's noise
        __builtin_prefetch(noise + base + HID + n0, 0, 0);
      }
    }

    // C[16x32] = h[16x256] @ W_eff^T[256x32], bf16 inputs, f32 accumulate.
    v8f c0 = {0.f,0.f,0.f,0.f,0.f,0.f,0.f,0.f};
    v8f c1 = c0;
    #pragma unroll
    for (int kf = 0; kf < 8; ++kf) {
      Frag a;
      const unsigned short* hr = &hRd[(lane & 15)*WPAD + kf*32 + aoff];
      a.q[0] = *(const uint4*)(hr);
      a.q[1] = *(const uint4*)(hr + 8);
      c0 = __builtin_amdgcn_wmma_f32_16x16x32_bf16(false, a.v, false, B0[kf].v,
                                                   (short)0, c0, false, false);
      c1 = __builtin_amdgcn_wmma_f32_16x16x32_bf16(false, a.v, false, B1[kf].v,
                                                   (short)0, c1, false, false);
    }

    // Epilogue: gate (HW v_tanh_f32), add noise, stream to HBM, write next-step
    // bf16 activations to the other hA buffer.
    #pragma unroll
    for (int v = 0; v < 8; ++v) {
      const float hv0 = htanh(c0[v] + xs[v]*wi0 + bh0) + nz0[v];
      const float hv1 = htanh(c1[v] + xs[v]*wi1 + bh1) + nz1[v];
      __builtin_nontemporal_store(hv0, outp + obase[v] + n0);
      __builtin_nontemporal_store(hv1, outp + obase[v] + n1);
      hWr[(moff + v)*WPAD + n0] = f2bf(hv0);
      hWr[(moff + v)*WPAD + n1] = f2bf(hv1);
    }
    __syncthreads();   // next-step h tile fully visible (double-buffered: one barrier/step)
  }
}

// Head: y[b,s] = sigmoid(out[b,s,:] . W[0,:] + b[0]).  One wave per (b,s).
// sigmoid(z) = 0.5 + 0.5*tanh(z/2): single TRANS op, no branches (HBM-bound anyway).
__global__ void head_kernel(const float* __restrict__ outp,
                            const float* __restrict__ W,
                            const float* __restrict__ b,
                            float* __restrict__ y)
{
  const int gw   = (blockIdx.x * blockDim.x + threadIdx.x) >> 5;
  const int lane = threadIdx.x & 31;
  if (gw >= BATCH * SEQ) return;
  const float* p = outp + (size_t)gw * HID;
  const float4 v0 = ((const float4*)p)[lane];
  const float4 v1 = ((const float4*)p)[lane + 32];
  const float4 w0 = ((const float4*)W)[lane];
  const float4 w1 = ((const float4*)W)[lane + 32];
  float acc = v0.x*w0.x + v0.y*w0.y + v0.z*w0.z + v0.w*w0.w
            + v1.x*w1.x + v1.y*w1.y + v1.z*w1.z + v1.w*w1.w;
  #pragma unroll
  for (int m = 16; m >= 1; m >>= 1) acc += __shfl_xor(acc, m, 32);
  if (lane == 0) y[gw] = 0.5f + 0.5f * htanh(0.5f * (acc + b[0]));
}

extern "C" void kernel_launch(void* const* d_in, const int* in_sizes, int n_in,
                              void* d_out, int out_size, void* d_ws, size_t ws_size,
                              hipStream_t stream) {
  (void)in_sizes; (void)n_in; (void)out_size; (void)d_ws; (void)ws_size;
  const float* x     = (const float*)d_in[0];
  const float* h0    = (const float*)d_in[1];
  const float* noise = (const float*)d_in[2];
  const float* W_ih  = (const float*)d_in[3];
  const float* W_hh  = (const float*)d_in[4];
  const float* W_hhb = (const float*)d_in[5];
  const float* b_h   = (const float*)d_in[6];
  const float* W     = (const float*)d_in[7];
  const float* b     = (const float*)d_in[8];
  const int*   ctx   = (const int*)d_in[9];

  float* y    = (float*)d_out;                          // (BATCH, SEQ)
  float* outp = (float*)d_out + (size_t)SEQ * BATCH;    // (BATCH, SEQ, HID)

  const size_t smem = (size_t)HID * WPAD * 2 + 2 * 16 * WPAD * 2;   // ~148.5 KB (<320 KB WGP LDS)
  rnn_scan_kernel<<<dim3(BATCH / 16), dim3(256), smem, stream>>>(
      x, h0, noise, W_ih, W_hh, W_hhb, b_h, ctx, outp);

  const int waves   = BATCH * SEQ;
  const int threads = 256;
  const int blocks  = (waves * 32 + threads - 1) / threads;
  head_kernel<<<dim3(blocks), dim3(threads), 0, stream>>>(outp, W, b, y);
}
